// LSTHM_81561428951471
// MI455X (gfx1250) — matrix operand order repaired and checked by reading, not compiled, not run on hardware
//
#include <hip/hip_runtime.h>
#include <stdint.h>

// ---------------------------------------------------------------------------
// LSTHM forward for MI455X (gfx1250), bf16 WMMA (v_wmma_f32_16x16x32_bf16).
// B=512, T=128, HID={128,32,32}, Z=64, HC=64, OC=64, OD=1.
// ---------------------------------------------------------------------------

typedef __attribute__((ext_vector_type(16))) __bf16 v16bf;
typedef __attribute__((ext_vector_type(8)))  float  v8f;

union FragU { unsigned int u[8]; v16bf v; };
union AccU  { float f[8]; v8f v; };

static __device__ __forceinline__ unsigned short f2bf(float f) {
    unsigned int x = __float_as_uint(f);
    x += 0x7FFFu + ((x >> 16) & 1u);           // round-to-nearest-even
    return (unsigned short)(x >> 16);
}
static __device__ __forceinline__ float bf2f(unsigned short h) {
    return __uint_as_float(((unsigned int)h) << 16);
}
static __device__ __forceinline__ float sigmoidf_(float x) {
    return 1.0f / (1.0f + __expf(-x));
}
static __device__ __forceinline__ v8f wmma_bf16(v16bf a, v16bf b, v8f c) {
    // 8 args: (neg_a, A, neg_b, B, c_mod, C, reuse_a, reuse_b)
    return __builtin_amdgcn_wmma_f32_16x16x32_bf16(false, a, false, b, (short)0, c, false, false);
}

// A-matrix 16x32 bf16 fragment (ISA 7.12.2): lane<16 -> row=lane, K in
// {k..k+7} U {k+16..k+23}; lane>=16 -> row=lane-16, K chunks shifted by 8.
static __device__ __forceinline__ void load_fragA(const unsigned short* rowbase,
                                                  int k, int rsel, FragU& fr) {
    const uint4* q0 = reinterpret_cast<const uint4*>(rowbase + k + rsel * 8);
    const uint4* q1 = reinterpret_cast<const uint4*>(rowbase + k + 16 + rsel * 8);
    uint4 x = *q0, y = *q1;
    fr.u[0] = x.x; fr.u[1] = x.y; fr.u[2] = x.z; fr.u[3] = x.w;
    fr.u[4] = y.x; fr.u[5] = y.y; fr.u[6] = y.z; fr.u[7] = y.w;
}
// B-matrix 32x16 bf16 fragment: lane<16 -> col=lane, K in {k..k+7} U
// {k+16..k+23}; lane>=16 -> col=lane-16, K chunks shifted by 8. colbase points at [N][Kpad] row start.
static __device__ __forceinline__ void load_fragB(const unsigned short* colbase,
                                                  int k, int rsel, FragU& fr) {
    const uint4* q0 = reinterpret_cast<const uint4*>(colbase + k + rsel * 8);
    const uint4* q1 = reinterpret_cast<const uint4*>(colbase + k + 16 + rsel * 8);
    uint4 x = *q0, y = *q1;
    fr.u[0] = x.x; fr.u[1] = x.y; fr.u[2] = x.z; fr.u[3] = x.w;
    fr.u[4] = y.x; fr.u[5] = y.y; fr.u[6] = y.z; fr.u[7] = y.w;
}

// ---------------------------------------------------------------------------
// Kernel 0a: transpose + convert weight [K][N] f32 -> [N][Kpad] bf16 (zero pad)
// ---------------------------------------------------------------------------
__global__ void transpose_bf16_kernel(const float* __restrict__ src,
                                      unsigned short* __restrict__ dst,
                                      int K, int N, int Kp) {
    int idx = blockIdx.x * blockDim.x + threadIdx.x;
    if (idx >= N * Kp) return;
    int n = idx / Kp, k = idx - n * Kp;
    dst[idx] = (k < K) ? f2bf(src[k * N + n]) : (unsigned short)0;
}

// Kernel 0b: combined bias  bcomb = W_b + U_b + V_b
__global__ void bias3_kernel(const float* __restrict__ a, const float* __restrict__ b,
                             const float* __restrict__ c, float* __restrict__ o, int n) {
    int i = blockIdx.x * blockDim.x + threadIdx.x;
    if (i < n) o[i] = a[i] + b[i] + c[i];
}

// ---------------------------------------------------------------------------
// Kernel 0c: pad+convert activations  x [R][d] f32 -> xp [R][Kp] bf16 (zeros)
// Removes all bounds checks from the projection GEMM's A-fragment loads.
// ---------------------------------------------------------------------------
__global__ __launch_bounds__(256) void pad_x_bf16_kernel(const float* __restrict__ x,
                                                         unsigned short* __restrict__ xp,
                                                         int d, int Kp, int R) {
    int idx = blockIdx.x * blockDim.x + threadIdx.x;
    if (idx >= R * Kp) return;
    int r = idx / Kp, k = idx - r * Kp;
    xp[idx] = (k < d) ? f2bf(x[(long)r * d + k]) : (unsigned short)0;
}

// ---------------------------------------------------------------------------
// Kernel 1: input projection  ia[t][b][:] = x[b][t][:] @ W + (Wb+Ub+Vb)
// One 16x16 tile per wave, 8 waves (=128 output cols) per block.
// xp is bf16 [B*T][Kp]; Wt is bf16 [4H][Kp]; ia is bf16 [T][B][4H].
// ---------------------------------------------------------------------------
__global__ __launch_bounds__(256) void proj_kernel(
    const unsigned short* __restrict__ xp, const unsigned short* __restrict__ Wt,
    const float* __restrict__ bcomb, unsigned short* __restrict__ ia,
    int Kp, int fourH) {
    const int tid  = threadIdx.x;
    const int lane = tid & 31;
    const int wave = tid >> 5;
    const int rlo  = lane & 15;
    const int rsel = lane >> 4;
    const int nb   = fourH >> 7;                   // 4H / 128
    const long mtile  = blockIdx.x / nb;
    const int  nchunk = blockIdx.x - (int)mtile * nb;
    const int  ncol   = nchunk * 128 + wave * 16 + rlo;
    const long r0     = mtile * 16;

    AccU acc;
    float bias = bcomb[ncol];
#pragma unroll
    for (int i = 0; i < 8; i++) acc.f[i] = bias;

    const unsigned short* xrow = xp + (r0 + rlo) * (long)Kp;   // A row
    const unsigned short* wcol = Wt + (long)ncol * Kp;         // B column

    for (int k = 0; k < Kp; k += 32) {
        FragU a, b;
        load_fragA(xrow, k, rsel, a);
        load_fragB(wcol, k, rsel, b);
        acc.v = wmma_bf16(a.v, b.v, acc.v);
    }
    // C layout: VGPR i, lanes<16 -> row i; lanes>=16 -> row i+8; col = lane&15.
#pragma unroll
    for (int i = 0; i < 8; i++) {
        long r = r0 + i + rsel * 8;
        long t = r & 127;              // T = 128
        long bb = r >> 7;
        ia[(t * 512 + bb) * (long)fourH + ncol] = f2bf(acc.f[i]);
    }
}

// ---------------------------------------------------------------------------
// Kernel 2: persistent per-batch-slice scan over T=128 steps + fused output.
// 16 batch rows per block; c in registers; h/z/a1 in LDS as bf16 A operands.
// Recurrent weights (<0.3 MB bf16 total) stream from L2 per step.
// ---------------------------------------------------------------------------
__global__ __launch_bounds__(256) void lsthm_scan_kernel(
    const unsigned short* __restrict__ ia0, const unsigned short* __restrict__ ia1,
    const unsigned short* __restrict__ ia2,
    const unsigned short* __restrict__ U0t, const unsigned short* __restrict__ U1t,
    const unsigned short* __restrict__ U2t,
    const unsigned short* __restrict__ V0t, const unsigned short* __restrict__ V1t,
    const unsigned short* __restrict__ V2t,
    const unsigned short* __restrict__ hy1t, const float* __restrict__ hy1b,
    const unsigned short* __restrict__ hy2t, const float* __restrict__ hy2b,
    const unsigned short* __restrict__ o1t,  const float* __restrict__ o1b,
    const float* __restrict__ o2w, const float* __restrict__ o2b,
    float* __restrict__ out) {
    __shared__ float          sS[16 * 768];   // gate pre-activations (f,i,o,g)x3
    __shared__ unsigned short sH[16 * 192];   // hcat (bf16)
    __shared__ unsigned short sZ[16 * 64];    // z    (bf16)
    __shared__ unsigned short sA1[16 * 64];   // relu(hcat@hy1) (bf16)

    const int tid  = threadIdx.x;
    const int lane = tid & 31;
    const int wave = tid >> 5;
    const int rlo  = lane & 15;
    const int rsel = lane >> 4;
    const int b0   = blockIdx.x * 16;

    float creg[12];
#pragma unroll
    for (int j = 0; j < 12; j++) creg[j] = 0.0f;
    for (int i = tid; i < 16 * 192; i += 256) sH[i] = 0;
    for (int i = tid; i < 16 * 64;  i += 256) sZ[i] = 0;
    __syncthreads();

    for (int t = 0; t < 128; t++) {
        // ---- phase 1: s = ia_t + h@U + z@V   (48 16x16 tiles over 8 waves)
        for (int tt = wave; tt < 48; tt += 8) {
            int Ku, hk0, fourH, scolbase, lt;
            const unsigned short *Ut, *Vt, *iab;
            if (tt < 32)      { lt = tt;      Ku = 128; hk0 = 0;   fourH = 512; scolbase = 0;   Ut = U0t; Vt = V0t; iab = ia0; }
            else if (tt < 40) { lt = tt - 32; Ku = 32;  hk0 = 128; fourH = 128; scolbase = 512; Ut = U1t; Vt = V1t; iab = ia1; }
            else              { lt = tt - 40; Ku = 32;  hk0 = 160; fourH = 128; scolbase = 640; Ut = U2t; Vt = V2t; iab = ia2; }
            const int ncol = lt * 16 + rlo;
            const unsigned short* iaT = iab + ((size_t)t * 512 + b0) * fourH;

            AccU acc;
#pragma unroll
            for (int i = 0; i < 8; i++)
                acc.f[i] = bf2f(iaT[(i + rsel * 8) * fourH + ncol]);

            for (int k = 0; k < Ku; k += 32) {          // h_prev @ U
                FragU a, b;
                load_fragA(&sH[rlo * 192 + hk0], k, rsel, a);
                load_fragB(&Ut[(size_t)ncol * Ku], k, rsel, b);
                acc.v = wmma_bf16(a.v, b.v, acc.v);
            }
            for (int k = 0; k < 64; k += 32) {          // z_prev @ V
                FragU a, b;
                load_fragA(&sZ[rlo * 64], k, rsel, a);
                load_fragB(&Vt[(size_t)ncol * 64], k, rsel, b);
                acc.v = wmma_bf16(a.v, b.v, acc.v);
            }
#pragma unroll
            for (int i = 0; i < 8; i++)
                sS[(i + rsel * 8) * 768 + scolbase + ncol] = acc.f[i];

            if (t + 1 < 128) {                           // warm L2/L0 for next step
                const unsigned short* nxt =
                    iab + ((size_t)(t + 1) * 512 + b0) * fourH + (size_t)rlo * fourH + ncol;
                __builtin_prefetch((const void*)nxt, 0, 1);
            }
        }
        __syncthreads();

        // ---- phase 2: gates + cell update (3072 (row,unit) pairs, 12/thread)
#pragma unroll
        for (int j = 0; j < 12; j++) {
            int p  = tid + 256 * j;
            int r  = p / 192;
            int gu = p - r * 192;
            int u, h, base;
            if (gu < 128)      { u = gu;       h = 128; base = 0;   }
            else if (gu < 160) { u = gu - 128; h = 32;  base = 512; }
            else               { u = gu - 160; h = 32;  base = 640; }
            const float* sr = &sS[r * 768 + base];
            float fg = sigmoidf_(sr[u]);
            float ig = sigmoidf_(sr[h + u]);
            float og = sigmoidf_(sr[2 * h + u]);
            float gg = tanhf(sr[3 * h + u]);
            float c  = fg * creg[j] + ig * gg;
            creg[j]  = c;
            sH[r * 192 + gu] = f2bf(tanhf(c) * og);
        }
        __syncthreads();

        // ---- phase 3: a1 = relu(hcat @ hy1 + b1)   (4 tiles, waves 0..3)
        if (wave < 4) {
            const int ncol = wave * 16 + rlo;
            AccU acc;
            float bias = hy1b[ncol];
#pragma unroll
            for (int i = 0; i < 8; i++) acc.f[i] = bias;
            for (int k = 0; k < 192; k += 32) {
                FragU a, b;
                load_fragA(&sH[rlo * 192], k, rsel, a);
                load_fragB(&hy1t[(size_t)ncol * 192], k, rsel, b);
                acc.v = wmma_bf16(a.v, b.v, acc.v);
            }
#pragma unroll
            for (int i = 0; i < 8; i++)
                sA1[(i + rsel * 8) * 64 + ncol] = f2bf(fmaxf(acc.f[i], 0.0f));
        }
        __syncthreads();

        // ---- phase 4: z = sigmoid(a1 @ hy2 + b2)   (4 tiles, waves 0..3)
        if (wave < 4) {
            const int ncol = wave * 16 + rlo;
            AccU acc;
            float bias = hy2b[ncol];
#pragma unroll
            for (int i = 0; i < 8; i++) acc.f[i] = bias;
            for (int k = 0; k < 64; k += 32) {
                FragU a, b;
                load_fragA(&sA1[rlo * 64], k, rsel, a);
                load_fragB(&hy2t[(size_t)ncol * 64], k, rsel, b);
                acc.v = wmma_bf16(a.v, b.v, acc.v);
            }
#pragma unroll
            for (int i = 0; i < 8; i++)
                sZ[(i + rsel * 8) * 64 + ncol] = f2bf(sigmoidf_(acc.f[i]));
        }
        __syncthreads();
    }

    // ---- fused output head: out = relu([hcat,z] @ o1 + b) @ o2 + b
    if (wave < 4) {
        const int ncol = wave * 16 + rlo;
        AccU acc;
        float bias = o1b[ncol];
#pragma unroll
        for (int i = 0; i < 8; i++) acc.f[i] = bias;
        for (int k = 0; k < 256; k += 32) {
            FragU a, b;
            if (k < 192) load_fragA(&sH[rlo * 192], k, rsel, a);
            else         load_fragA(&sZ[rlo * 64], k - 192, rsel, a);
            load_fragB(&o1t[(size_t)ncol * 256], k, rsel, b);
            acc.v = wmma_bf16(a.v, b.v, acc.v);
        }
#pragma unroll
        for (int i = 0; i < 8; i++)
            sS[(i + rsel * 8) * 64 + ncol] = fmaxf(acc.f[i], 0.0f);
    }
    __syncthreads();
    if (tid < 16) {
        float s = o2b[0];
#pragma unroll 8
        for (int j = 0; j < 64; j++) s += sS[tid * 64 + j] * o2w[j];
        out[b0 + tid] = s;
    }
}

// ---------------------------------------------------------------------------
// Host-side launch
// ---------------------------------------------------------------------------
extern "C" void kernel_launch(void* const* d_in, const int* in_sizes, int n_in,
                              void* d_out, int out_size, void* d_ws, size_t ws_size,
                              hipStream_t stream) {
    (void)in_sizes; (void)n_in; (void)out_size; (void)ws_size;
    const float* x0 = (const float*)d_in[0];
    const float* x1 = (const float*)d_in[1];
    const float* x2 = (const float*)d_in[2];
    const float *W0w = (const float*)d_in[3],  *W0b = (const float*)d_in[4];
    const float *U0w = (const float*)d_in[5],  *U0b = (const float*)d_in[6];
    const float *V0w = (const float*)d_in[7],  *V0b = (const float*)d_in[8];
    const float *W1w = (const float*)d_in[9],  *W1b = (const float*)d_in[10];
    const float *U1w = (const float*)d_in[11], *U1b = (const float*)d_in[12];
    const float *V1w = (const float*)d_in[13], *V1b = (const float*)d_in[14];
    const float *W2w = (const float*)d_in[15], *W2b = (const float*)d_in[16];
    const float *U2w = (const float*)d_in[17], *U2b = (const float*)d_in[18];
    const float *V2w = (const float*)d_in[19], *V2b = (const float*)d_in[20];
    const float *hy1w = (const float*)d_in[21], *hy1b = (const float*)d_in[22];
    const float *hy2w = (const float*)d_in[23], *hy2b = (const float*)d_in[24];
    const float *o1w  = (const float*)d_in[25], *o1b  = (const float*)d_in[26];
    const float *o2w  = (const float*)d_in[27], *o2b  = (const float*)d_in[28];
    float* out = (float*)d_out;

    char* base = (char*)d_ws;
    size_t off = 0;
    auto a_u16 = [&](size_t n) -> unsigned short* {
        unsigned short* p = (unsigned short*)(base + off);
        off = (off + n * 2 + 255) & ~(size_t)255;
        return p;
    };
    auto a_f32 = [&](size_t n) -> float* {
        float* p = (float*)(base + off);
        off = (off + n * 4 + 255) & ~(size_t)255;
        return p;
    };

    unsigned short* W0t = a_u16(512 * 320);
    unsigned short* W1t = a_u16(128 * 96);
    unsigned short* W2t = a_u16(128 * 64);
    unsigned short* U0t = a_u16(512 * 128);
    unsigned short* U1t = a_u16(128 * 32);
    unsigned short* U2t = a_u16(128 * 32);
    unsigned short* V0t = a_u16(512 * 64);
    unsigned short* V1t = a_u16(128 * 64);
    unsigned short* V2t = a_u16(128 * 64);
    unsigned short* hy1t = a_u16(64 * 192);
    unsigned short* hy2t = a_u16(64 * 64);
    unsigned short* o1t  = a_u16(64 * 256);
    float* bc0 = a_f32(512);
    float* bc1 = a_f32(128);
    float* bc2 = a_f32(128);
    unsigned short* ia0 = a_u16((size_t)128 * 512 * 512);
    unsigned short* ia1 = a_u16((size_t)128 * 512 * 128);
    unsigned short* ia2 = a_u16((size_t)128 * 512 * 128);
    unsigned short* xp0 = a_u16((size_t)65536 * 320);
    unsigned short* xp1 = a_u16((size_t)65536 * 96);
    unsigned short* xp2 = a_u16((size_t)65536 * 64);

    auto tr = [&](const float* src, unsigned short* dst, int K, int N, int Kp) {
        int total = N * Kp;
        transpose_bf16_kernel<<<(total + 255) / 256, 256, 0, stream>>>(src, dst, K, N, Kp);
    };
    tr(W0w, W0t, 300, 512, 320);  tr(W1w, W1t, 74, 128, 96);  tr(W2w, W2t, 35, 128, 64);
    tr(U0w, U0t, 128, 512, 128);  tr(U1w, U1t, 32, 128, 32);  tr(U2w, U2t, 32, 128, 32);
    tr(V0w, V0t, 64, 512, 64);    tr(V1w, V1t, 64, 128, 64);  tr(V2w, V2t, 64, 128, 64);
    tr(hy1w, hy1t, 192, 64, 192); tr(hy2w, hy2t, 64, 64, 64); tr(o1w, o1t, 256, 64, 256);

    bias3_kernel<<<2, 256, 0, stream>>>(W0b, U0b, V0b, bc0, 512);
    bias3_kernel<<<1, 128, 0, stream>>>(W1b, U1b, V1b, bc1, 128);
    bias3_kernel<<<1, 128, 0, stream>>>(W2b, U2b, V2b, bc2, 128);

    // pad+convert activations to bf16 (branch-free WMMA A operands)
    const int R = 65536;   // B*T
    pad_x_bf16_kernel<<<((size_t)R * 320 + 255) / 256, 256, 0, stream>>>(x0, xp0, 300, 320, R);
    pad_x_bf16_kernel<<<((size_t)R * 96  + 255) / 256, 256, 0, stream>>>(x1, xp1, 74,  96,  R);
    pad_x_bf16_kernel<<<((size_t)R * 64  + 255) / 256, 256, 0, stream>>>(x2, xp2, 35,  64,  R);

    // 65536 rows / 16 = 4096 M-tiles; 128 output cols per block.
    proj_kernel<<<4096 * 4, 256, 0, stream>>>(xp0, W0t, bc0, ia0, 320, 512);
    proj_kernel<<<4096 * 1, 256, 0, stream>>>(xp1, W1t, bc1, ia1, 96,  128);
    proj_kernel<<<4096 * 1, 256, 0, stream>>>(xp2, W2t, bc2, ia2, 64,  128);

    lsthm_scan_kernel<<<32, 256, 0, stream>>>(ia0, ia1, ia2,
                                              U0t, U1t, U2t, V0t, V1t, V2t,
                                              hy1t, hy1b, hy2t, hy2b,
                                              o1t, o1b, o2w, o2b, out);
}
